// LatentDenseGCN_7851200217644
// MI455X (gfx1250) — matrix operand order repaired and checked by reading, not compiled, not run on hardware
//
#include <hip/hip_runtime.h>
#include <hip/hip_bf16.h>
#include <math.h>

// ---------------------------------------------------------------------------
// LatentDenseGCN fused kernel for gfx1250 (MI455X), bf16 WMMA + f32 accum.
//
// Compute-bound (~120 GFLOP dense matmul vs 67MB mandatory traffic), so all
// GEMMs go through v_wmma_f32_16x16x32_bf16. normA lives in LDS (128KB bf16),
// H ping-pongs in workspace (L2-resident), weights are pre-swizzled into the
// WMMA B-operand lane layout so B loads are contiguous 32B per lane.
//
// Workspace layout (bytes), total ~68.4 MB:
//   [0)          xw1p  : 256*512 bf16 (X@w1+b1, B-swizzled)      =   256 KB
//   [256K)       w2p   : 512*512 bf16 (B-swizzled)               =   512 KB
//   [768K)       w3p   : 512*512 bf16 (B-swizzled)               =   512 KB
//   [1280K)      wsH   : 128 WGs * 2 * 256*512 bf16 slots        =    64 MB
// ---------------------------------------------------------------------------

typedef __bf16 bf16_t;
typedef __attribute__((ext_vector_type(16))) __bf16 v16bf;
typedef __attribute__((ext_vector_type(8)))  __bf16 v8bf;
typedef __attribute__((ext_vector_type(8)))  float   v8f;

#define NUM_NODES 256
#define IN_FEAT   64
#define HIDDEN    512
#define NCLS      2
#define BATCH     256
#define MAIN_GRID 128   // persistent WGs; each handles BATCH/MAIN_GRID batches

// ---- WMMA B-operand swizzle -----------------------------------------------
// B (K x N) element (k,n) lives at: tile (k/32, n/16), lane = (n&15)|(((k>>4)&1)<<4),
// j = k&15. Per (tile,lane) the 16 bf16 values are contiguous (32 bytes).
__device__ __forceinline__ size_t packB_idx(int k, int n, int ncols) {
  size_t kt   = (size_t)(k >> 5);
  size_t nt   = (size_t)(n >> 4);
  size_t lane = (size_t)((n & 15) | (((k >> 4) & 1) << 4));
  return (((kt * (size_t)(ncols >> 4) + nt) << 5) + lane) * 16 + (size_t)(k & 15);
}

// ---- operand loaders -------------------------------------------------------
// A operand (16x32 bf16) from a row-major matrix: lane m = lane&15,
// kbase = (lane&16)?16:0 ; a[j] = A[m][kb+j].
__device__ __forceinline__ v16bf load_a_rm(const bf16_t* base, int ld, int lane) {
  int m  = lane & 15;
  int kb = (lane & 16) ? 16 : 0;
  const bf16_t* r = base + (size_t)m * ld + kb;
  v8bf lo = *(const v8bf*)r;          // K = kb .. kb+7
  v8bf hi = *(const v8bf*)(r + 8);    // K = kb+8 .. kb+15
  v16bf out;
#pragma unroll
  for (int j = 0; j < 8; ++j) { out[j] = lo[j]; out[j + 8] = hi[j]; }
  return out;
}

// B operand: 16 contiguous bf16 (32 bytes) at p (pre-swizzled layout / Tt rows).
__device__ __forceinline__ v16bf load_b_contig(const bf16_t* p) {
  v8bf lo = *(const v8bf*)p;
  v8bf hi = *(const v8bf*)(p + 8);
  v16bf out;
#pragma unroll
  for (int j = 0; j < 8; ++j) { out[j] = lo[j]; out[j + 8] = hi[j]; }
  return out;
}

__device__ __forceinline__ v8f wmma_bf16(v16bf a, v16bf b, v8f c) {
  // 8 args: (neg_a, A, neg_b, B, c_mod, C, reuse_a, reuse_b)
  return __builtin_amdgcn_wmma_f32_16x16x32_bf16(false, a, false, b, (short)0, c,
                                                 false, false);
}

// D tile (16x16 f32) -> relu -> bf16 row-major store.
// D layout: vgpr g, lanes 0-15: M=m0+g, N=n0+lane; lanes 16-31: M=m0+8+g.
__device__ __forceinline__ void store_relu_rm(bf16_t* H, int ld, int m0, int n0,
                                              int lane, v8f c) {
  int n     = n0 + (lane & 15);
  int mbase = m0 + ((lane & 16) ? 8 : 0);
#pragma unroll
  for (int g = 0; g < 8; ++g) {
    float v = c[g];
    v = v > 0.f ? v : 0.f;
    H[(size_t)(mbase + g) * ld + n] = (bf16_t)v;
  }
}

// D tile + bias -> transposed bf16 store into Tt[n][m] (row stride 256):
// per lane the 8 m-values are contiguous -> one 16B LDS store.
__device__ __forceinline__ void store_Tt(bf16_t* Tt, int mt, int ntl, int lane,
                                         v8f c, float bv) {
  int n     = ntl * 16 + (lane & 15);
  int mbase = mt * 16 + ((lane & 16) ? 8 : 0);
  v8bf o;
#pragma unroll
  for (int g = 0; g < 8; ++g) o[g] = (bf16_t)(c[g] + bv);
  *(v8bf*)(Tt + (size_t)n * NUM_NODES + mbase) = o;
}

// ---- prep kernels ----------------------------------------------------------
__global__ void prep_xw1_kernel(const float* __restrict__ X, const float* __restrict__ w1,
                                const float* __restrict__ b1, bf16_t* __restrict__ out) {
  int idx = blockIdx.x * blockDim.x + threadIdx.x;  // 256*512 threads
  int r = idx >> 9, c = idx & (HIDDEN - 1);
  float s = b1[c];
#pragma unroll
  for (int f = 0; f < IN_FEAT; ++f) s += X[r * IN_FEAT + f] * w1[f * HIDDEN + c];
  out[packB_idx(r, c, HIDDEN)] = (bf16_t)s;
}

__global__ void prep_pack_kernel(const float* __restrict__ W, bf16_t* __restrict__ out) {
  int idx = blockIdx.x * blockDim.x + threadIdx.x;  // 512*512 threads
  int k = idx >> 9, n = idx & (HIDDEN - 1);
  out[packB_idx(k, n, HIDDEN)] = (bf16_t)W[idx];
}

// ---- main fused kernel -----------------------------------------------------
__global__ __launch_bounds__(256) void gcn_fused_kernel(
    const float* __restrict__ z, const float* __restrict__ b2v,
    const float* __restrict__ b3v, const float* __restrict__ fcw,
    const float* __restrict__ fcb, const bf16_t* __restrict__ xw1p,
    const bf16_t* __restrict__ w2p, const bf16_t* __restrict__ w3p,
    bf16_t* __restrict__ wsH, float* __restrict__ out) {
  extern __shared__ char smem_raw[];
  bf16_t* sA  = (bf16_t*)smem_raw;                   // 256*256 bf16 = 128 KB
  bf16_t* sT  = sA + NUM_NODES * NUM_NODES;          // 32*256  bf16 =  16 KB (T^t staging)
  float*  sD  = (float*)(sT + 32 * NUM_NODES);       // 256 f32 (d^{-1/2})
  float*  sGe = sD + NUM_NODES;                      // 512 f32 (graph emb)

  const int tid  = threadIdx.x;
  const int wave = tid >> 5;
  const int lane = tid & 31;

  bf16_t* slot0 = wsH + (size_t)blockIdx.x * 2 * (NUM_NODES * HIDDEN);
  bf16_t* slot1 = slot0 + NUM_NODES * HIDDEN;

  for (int b = blockIdx.x; b < BATCH; b += gridDim.x) {
    const float* zb = z + (size_t)b * NUM_NODES * NUM_NODES;

    // ---- Phase A: normA = D^-1/2 (sym(sigmoid(z)) + I) D^-1/2, bf16 in LDS
    for (int idx = tid; idx < NUM_NODES * NUM_NODES; idx += blockDim.x) {
      int i = idx >> 8, j = idx & 255;
      float zij = zb[idx];
      float zji = zb[j * NUM_NODES + i];
      float a = 0.5f * (1.f / (1.f + __expf(-zij)) + 1.f / (1.f + __expf(-zji)));
      if (i == j) a += 1.0f;
      sA[idx] = (bf16_t)a;
    }
    __syncthreads();
    if (tid < NUM_NODES) {
      float s = 1e-6f;
      for (int j = 0; j < NUM_NODES; ++j) s += (float)sA[tid * NUM_NODES + j];
      sD[tid] = rsqrtf(s);
    }
    __syncthreads();
    for (int idx = tid; idx < NUM_NODES * NUM_NODES; idx += blockDim.x) {
      int i = idx >> 8, j = idx & 255;
      sA[idx] = (bf16_t)((float)sA[idx] * sD[i] * sD[j]);
    }
    __syncthreads();

    // ---- Layer 1: H1 = relu(normA @ (X@w1+b1))  [M=256,K=256,N=512]
    for (int t = wave; t < 16 * 32; t += 8) {
      int mt = t >> 5, nt = t & 31;
      v8f c = {};
#pragma unroll
      for (int kt = 0; kt < 8; ++kt) {
        v16bf a  = load_a_rm(sA + (size_t)mt * 16 * NUM_NODES + kt * 32, NUM_NODES, lane);
        v16bf bm = load_b_contig(xw1p + (((size_t)kt * 32 + nt) * 32 + lane) * 16);
        c = wmma_bf16(a, bm, c);
      }
      store_relu_rm(slot0, HIDDEN, mt * 16, nt * 16, lane, c);
    }
    __threadfence();
    __syncthreads();

    // ---- Layers 2,3: H <- relu(normA @ (H@W + b)), 32 cols at a time
    const bf16_t* Hin  = slot0;
    bf16_t*       Hout = slot1;
#pragma unroll 1
    for (int layer = 0; layer < 2; ++layer) {
      const bf16_t* Wp   = (layer == 0) ? w2p : w3p;
      const float*  bias = (layer == 0) ? b2v : b3v;
      for (int hb = 0; hb < 16; ++hb) {
        // T^t = (Hin @ Wp[:, hb*32 .. +32) + bias)^T into LDS  [M=256,K=512,N=32]
        for (int t = wave; t < 32; t += 8) {
          int mt = t >> 1, ntl = t & 1;
          v8f c = {};
#pragma unroll
          for (int kt = 0; kt < 16; ++kt) {
            v16bf a  = load_a_rm(Hin + (size_t)mt * 16 * HIDDEN + kt * 32, HIDDEN, lane);
            v16bf bm = load_b_contig(Wp + (((size_t)kt * 32 + (hb * 2 + ntl)) * 32 + lane) * 16);
            c = wmma_bf16(a, bm, c);
          }
          float bv = bias[hb * 32 + ntl * 16 + (lane & 15)];
          store_Tt(sT, mt, ntl, lane, c, bv);
        }
        __syncthreads();
        // Hout[:, hb block] = relu(normA @ T)  [M=256,K=256,N=32]
        for (int t = wave; t < 32; t += 8) {
          int mt = t >> 1, ntl = t & 1;
          v8f c = {};
#pragma unroll
          for (int kt = 0; kt < 8; ++kt) {
            v16bf a  = load_a_rm(sA + (size_t)mt * 16 * NUM_NODES + kt * 32, NUM_NODES, lane);
            // B from Tt: b[j] = Tt[n][kt*32 + kbase + j], kbase = (lane&16)?16:0
            const bf16_t* tp = sT + (size_t)(ntl * 16 + (lane & 15)) * NUM_NODES +
                               kt * 32 + ((lane & 16) ? 16 : 0);
            v16bf bm = load_b_contig(tp);
            c = wmma_bf16(a, bm, c);
          }
          store_relu_rm(Hout, HIDDEN, mt * 16, (hb * 2 + ntl) * 16, lane, c);
        }
        __syncthreads();
      }
      __threadfence();
      __syncthreads();
      const bf16_t* tmp = Hin; Hin = Hout; Hout = (bf16_t*)tmp;
    }
    // H3 now in Hin (== slot0)

    // ---- mean pool + classifier
    for (int h = tid; h < HIDDEN; h += blockDim.x) {
      float s = 0.f;
      for (int m = 0; m < NUM_NODES; ++m) s += (float)Hin[(size_t)m * HIDDEN + h];
      sGe[h] = s * (1.0f / NUM_NODES);
    }
    __syncthreads();
    if (tid < NCLS) {
      float s = fcb[tid];
      for (int h = 0; h < HIDDEN; ++h) s += sGe[h] * fcw[h * NCLS + tid];
      out[b * NCLS + tid] = s;
    }
    __syncthreads();
  }
}

// ---------------------------------------------------------------------------
extern "C" void kernel_launch(void* const* d_in, const int* in_sizes, int n_in,
                              void* d_out, int out_size, void* d_ws, size_t ws_size,
                              hipStream_t stream) {
  const float* z   = (const float*)d_in[0];
  const float* ne  = (const float*)d_in[1];
  const float* w1  = (const float*)d_in[2];
  const float* b1  = (const float*)d_in[3];
  const float* w2  = (const float*)d_in[4];
  const float* b2  = (const float*)d_in[5];
  const float* w3  = (const float*)d_in[6];
  const float* b3  = (const float*)d_in[7];
  const float* fcw = (const float*)d_in[8];
  const float* fcb = (const float*)d_in[9];
  float*       out = (float*)d_out;

  bf16_t* xw1p = (bf16_t*)d_ws;                 // 256*512
  bf16_t* w2p  = xw1p + NUM_NODES * HIDDEN;     // 512*512
  bf16_t* w3p  = w2p + HIDDEN * HIDDEN;         // 512*512
  bf16_t* wsH  = w3p + HIDDEN * HIDDEN;         // MAIN_GRID * 2 * 256*512

  prep_xw1_kernel<<<(NUM_NODES * HIDDEN) / 256, 256, 0, stream>>>(ne, w1, b1, xw1p);
  prep_pack_kernel<<<(HIDDEN * HIDDEN) / 256, 256, 0, stream>>>(w2, w2p);
  prep_pack_kernel<<<(HIDDEN * HIDDEN) / 256, 256, 0, stream>>>(w3, w3p);

  size_t smem = (size_t)(NUM_NODES * NUM_NODES + 32 * NUM_NODES) * sizeof(bf16_t) +
                (size_t)(NUM_NODES + HIDDEN) * sizeof(float);  // 150528 B
  gcn_fused_kernel<<<MAIN_GRID, 256, smem, stream>>>(z, b2, b3, fcw, fcb, xw1p, w2p,
                                                     w3p, wsH, out);
}
